// Linear_Transformer_60430189855383
// MI455X (gfx1250) — matrix operand order repaired and checked by reading, not compiled, not run on hardware
//
#include <hip/hip_runtime.h>
#include <stdint.h>

typedef unsigned short u16;
typedef unsigned int   u32;
typedef __attribute__((ext_vector_type(16))) __bf16 v16bf;
typedef __attribute__((ext_vector_type(8)))  float  v8f;

#define NB      8
#define NTOK    3000
#define MAXSEQ  4096
#define DIN     128
#define DM      512
#define DH      128
#define DFF     2048
#define DOUTN   64
#define ROWS    (NB*MAXSEQ)   /* 32768 */
#define NWIN    (MAXSEQ/128)  /* 32 */

// ---------------- bf16 helpers (RNE) ----------------
__device__ __forceinline__ u16 f2bf(float x){
  u32 u = __builtin_bit_cast(u32, x);
  if ((u & 0x7fffffffu) > 0x7f800000u) return (u16)((u >> 16) | 0x0040u); // qNaN
  u += 0x7fffu + ((u >> 16) & 1u);
  return (u16)(u >> 16);
}
__device__ __forceinline__ float bf2f(u16 h){ return __builtin_bit_cast(float, (u32)h << 16); }

union FragBF { v16bf v; uint4 q[2]; };

#define WMMA_BF16(ACC, A, B) \
  __builtin_amdgcn_wmma_f32_16x16x32_bf16(false, (A), false, (B), (short)0, (ACC), false, false)

// A-operand (16x32) per-lane chunks: lanes0-15 K={0..7,16..23}, lanes16-31 K={8..15,24..31}
// B-operand (32x16) per-lane chunks: lanes0-15 K={0..7,16..23}, lanes16-31 K={8..15,24..31}
// C/D: VGPR r -> M=r (lanes0-15) / M=r+8 (lanes16-31); N = lane&15.

// Fragment set for one K-32 step of a 32Mx64N wave tile.
struct TileFrags { FragBF a[2]; FragBF b[4]; };

__device__ __forceinline__ void load_frags(
    TileFrags& t, const u16* pa0, const u16* pa1,
    const u16* pb0, const u16* pb1, const u16* pb2, const u16* pb3,
    int k, int half)
{
  const int a0 = k + half*8, a1 = k + 16 + half*8;
  const int b0 = k + half*8, b1 = b0 + 16;
  t.a[0].q[0] = *(const uint4*)(pa0 + a0);
  t.a[0].q[1] = *(const uint4*)(pa0 + a1);
  t.a[1].q[0] = *(const uint4*)(pa1 + a0);
  t.a[1].q[1] = *(const uint4*)(pa1 + a1);
  t.b[0].q[0] = *(const uint4*)(pb0 + b0);
  t.b[0].q[1] = *(const uint4*)(pb0 + b1);
  t.b[1].q[0] = *(const uint4*)(pb1 + b0);
  t.b[1].q[1] = *(const uint4*)(pb1 + b1);
  t.b[2].q[0] = *(const uint4*)(pb2 + b0);
  t.b[2].q[1] = *(const uint4*)(pb2 + b1);
  t.b[3].q[0] = *(const uint4*)(pb3 + b0);
  t.b[3].q[1] = *(const uint4*)(pb3 + b1);
}

// ---------------- generic batched WMMA GEMM ----------------
// A: bf16 [M][K] row-major (lda). B: bf16 [N][K] row-major (ldb, "pre-transposed").
// C: f32 or bf16, [M][N] (ldc). flags: 1=store bf16, 2=accumulate into f32 C,
// 4=tanh, 8=gelu(exact). Requires K % 64 == 0, M % 32 == 0, N % 64 == 0.
// Wave tile = 32M x 64N, software-pipelined (double-buffered fragments).
// grid=(ceil(M/128), N/64, nZb*nZh), block=128.
__global__ void __launch_bounds__(128) gemm_wmma(
    const u16* __restrict__ A, long long sAb, long long sAh, int lda,
    const u16* __restrict__ B, long long sBb, long long sBh, int ldb,
    void* __restrict__ C, long long sCb, long long sCh, int ldc,
    const float* __restrict__ bias, int M, int N, int K, int nZh, int flags)
{
  const int z  = blockIdx.z;
  const int zb = z / nZh, zh = z % nZh;
  A += zb*sAb + zh*sAh;
  B += zb*sBb + zh*sBh;
  const long long cOff = zb*sCb + zh*sCh;

  const int wave = threadIdx.x >> 5, lane = threadIdx.x & 31;
  const int half = lane >> 4,  l16 = lane & 15;
  const int m0 = (blockIdx.x * 4 + wave) * 32;
  if (m0 >= M) return;
  const int n0 = blockIdx.y * 64;

  const u16* pa0 = A + (long long)(m0 + l16) * lda;
  const u16* pa1 = pa0 + (long long)16 * lda;
  const u16* pb0 = B + (long long)(n0 + l16) * ldb;
  const u16* pb1 = pb0 + (long long)16 * ldb;
  const u16* pb2 = pb0 + (long long)32 * ldb;
  const u16* pb3 = pb0 + (long long)48 * ldb;

  v8f acc[2][4] = {};
  TileFrags t0, t1;

  load_frags(t0, pa0, pa1, pb0, pb1, pb2, pb3, 0, half);
  int k = 0;
  for (; k + 64 < K; k += 64) {
    load_frags(t1, pa0, pa1, pb0, pb1, pb2, pb3, k + 32, half);
#pragma unroll
    for (int mi = 0; mi < 2; ++mi)
#pragma unroll
      for (int s = 0; s < 4; ++s)
        acc[mi][s] = WMMA_BF16(acc[mi][s], t0.a[mi].v, t0.b[s].v);
    load_frags(t0, pa0, pa1, pb0, pb1, pb2, pb3, k + 64, half);
#pragma unroll
    for (int mi = 0; mi < 2; ++mi)
#pragma unroll
      for (int s = 0; s < 4; ++s)
        acc[mi][s] = WMMA_BF16(acc[mi][s], t1.a[mi].v, t1.b[s].v);
  }
  load_frags(t1, pa0, pa1, pb0, pb1, pb2, pb3, k + 32, half);
#pragma unroll
  for (int mi = 0; mi < 2; ++mi)
#pragma unroll
    for (int s = 0; s < 4; ++s)
      acc[mi][s] = WMMA_BF16(acc[mi][s], t0.a[mi].v, t0.b[s].v);
#pragma unroll
  for (int mi = 0; mi < 2; ++mi)
#pragma unroll
    for (int s = 0; s < 4; ++s)
      acc[mi][s] = WMMA_BF16(acc[mi][s], t1.a[mi].v, t1.b[s].v);

  const bool stbf  = flags & 1, accum = flags & 2;
  const bool dtanh = flags & 4, dgelu = flags & 8;
#pragma unroll
  for (int mi = 0; mi < 2; ++mi) {
#pragma unroll
    for (int s = 0; s < 4; ++s) {
      const int col = n0 + s*16 + l16;
      const float bi = bias ? bias[col] : 0.f;
#pragma unroll
      for (int r = 0; r < 8; ++r) {
        const int m = m0 + mi*16 + r + 8*half;
        float v = acc[mi][s][r] + bi;
        if (dtanh) v = tanhf(v);
        if (dgelu) v = 0.5f * v * (1.f + erff(v * 0.70710678118654752f));
        if (stbf) {
          ((u16*)C)[cOff + (long long)m*ldc + col] = f2bf(v);
        } else {
          float* cp = (float*)C + cOff + (long long)m*ldc + col;
          if (accum) v += *cp;
          *cp = v;
        }
      }
    }
  }
}

// ---------------- pad + convert x ----------------
__global__ void pad_x_kernel(const float* __restrict__ x, u16* __restrict__ Xb){
  long i = (long)blockIdx.x * 256 + threadIdx.x;          // over NB*4096*128
  if (i >= (long)NB*MAXSEQ*DIN) return;
  int c = (int)(i % DIN); long t = i / DIN;
  int r = (int)(t % MAXSEQ); int b = (int)(t / MAXSEQ);
  Xb[i] = (r < NTOK) ? f2bf(x[((long)b*NTOK + r)*DIN + c]) : (u16)0;
}

// ---------------- weight transpose-convert: W[K][N] f32 -> Wt[N][K] bf16 ----------------
__global__ void wconv_kernel(const float* __restrict__ W, u16* __restrict__ Wt, int K, int N){
  long i = (long)blockIdx.x * 256 + threadIdx.x;
  if (i >= (long)K*N) return;
  int n = (int)(i / K), k = (int)(i % K);
  Wt[i] = f2bf(W[(long)k*N + n]);
}

// ---------------- layernorm (wave per row) -> bf16 ----------------
__global__ void __launch_bounds__(256) ln_kernel(
    const float* __restrict__ X, const float* __restrict__ g,
    const float* __restrict__ bta, u16* __restrict__ out)
{
  const int row = blockIdx.x * 8 + (threadIdx.x >> 5);
  const int lane = threadIdx.x & 31;
  const float* x = X + (long long)row * DM;
  float4 v[4]; float s = 0.f, s2 = 0.f;
#pragma unroll
  for (int i = 0; i < 4; ++i) {
    v[i] = *(const float4*)(x + i*128 + lane*4);
    s  += v[i].x + v[i].y + v[i].z + v[i].w;
    s2 += v[i].x*v[i].x + v[i].y*v[i].y + v[i].z*v[i].z + v[i].w*v[i].w;
  }
#pragma unroll
  for (int o = 16; o >= 1; o >>= 1) { s += __shfl_xor(s, o); s2 += __shfl_xor(s2, o); }
  const float mean = s * (1.f/512.f);
  const float var  = s2 * (1.f/512.f) - mean*mean;
  const float rstd = rsqrtf(var + 1e-5f);
  u16* op = out + (long long)row * DM;
#pragma unroll
  for (int i = 0; i < 4; ++i) {
    const int idx = i*128 + lane*4;
    float r0 = (v[i].x - mean)*rstd*g[idx+0] + bta[idx+0];
    float r1 = (v[i].y - mean)*rstd*g[idx+1] + bta[idx+1];
    float r2 = (v[i].z - mean)*rstd*g[idx+2] + bta[idx+2];
    float r3 = (v[i].w - mean)*rstd*g[idx+3] + bta[idx+3];
    uint2 pk; pk.x = (u32)f2bf(r0) | ((u32)f2bf(r1) << 16);
    pk.y = (u32)f2bf(r2) | ((u32)f2bf(r3) << 16);
    *(uint2*)(op + idx) = pk;
  }
}

// ---------------- q row-softmax prep (global heads 1..3) ----------------
__global__ void __launch_bounds__(96) qprep_kernel(const u16* __restrict__ Q, u16* __restrict__ QP){
  const int row = blockIdx.x;
  const int h = threadIdx.x >> 5, lane = threadIdx.x & 31;
  const u16* src = Q + (long long)row*DM + DH + h*DH + lane*4;
  uint2 u = *(const uint2*)src;
  float x0 = bf2f((u16)(u.x & 0xffff)), x1 = bf2f((u16)(u.x >> 16));
  float x2 = bf2f((u16)(u.y & 0xffff)), x3 = bf2f((u16)(u.y >> 16));
  float m = fmaxf(fmaxf(x0,x1), fmaxf(x2,x3));
#pragma unroll
  for (int o = 16; o >= 1; o >>= 1) m = fmaxf(m, __shfl_xor(m, o));
  float e0 = expf(x0-m), e1 = expf(x1-m), e2 = expf(x2-m), e3 = expf(x3-m);
  float s = e0+e1+e2+e3;
#pragma unroll
  for (int o = 16; o >= 1; o >>= 1) s += __shfl_xor(s, o);
  const float inv = 0.08838834764831845f / s;  // * dh^-0.5
  uint2 pk; pk.x = (u32)f2bf(e0*inv) | ((u32)f2bf(e1*inv) << 16);
  pk.y = (u32)f2bf(e2*inv) | ((u32)f2bf(e3*inv) << 16);
  *(uint2*)(QP + (long long)row*384 + h*DH + lane*4) = pk;
}

// ---------------- k column softmax stats (per b,h': max & sumexp over tokens) ----------------
__global__ void __launch_bounds__(256) kstats_kernel(
    const u16* __restrict__ K, float* __restrict__ cmax, float* __restrict__ csum)
{
  const int z = blockIdx.x;            // b*3 + h'
  const int b = z / 3, h = z % 3;
  const int d = threadIdx.x & 127, seg = threadIdx.x >> 7;
  const u16* base = K + (long long)b*MAXSEQ*DM + DH + h*DH + d;
  float m = -INFINITY, s = 0.f;
  for (int n = seg; n < MAXSEQ; n += 2) {
    float x = bf2f(base[(long long)n*DM]);
    float nm = fmaxf(m, x);
    s = s*expf(m-nm) + expf(x-nm);
    m = nm;
  }
  __shared__ float sm[256], ss[256];
  sm[threadIdx.x] = m; ss[threadIdx.x] = s;
  __syncthreads();
  if (seg == 0) {
    float m1 = sm[128+d], s1 = ss[128+d];
    float nm = fmaxf(m, m1);
    float S  = s*expf(m-nm) + s1*expf(m1-nm);
    cmax[z*128 + d] = nm; csum[z*128 + d] = S;
  }
}

// ---------------- tiled transpose: in[b][4096][512] bf16 -> out[b][outRows][4096] bf16 ----
// out[b][c][n] = f(in[b][n][colOff + c]); optional exp-normalize (linear-attn k'')
__global__ void __launch_bounds__(256) transpose_kernel(
    const u16* __restrict__ in, u16* __restrict__ out,
    const float* __restrict__ cmax, const float* __restrict__ csum,
    int colOff, int outRows)
{
  __shared__ u16 tile[32][33];
  const int b  = blockIdx.z;
  const int n0 = blockIdx.x * 32, c0 = blockIdx.y * 32;
  const int tx = threadIdx.x & 31, ty = threadIdx.x >> 5;
  const u16* src = in + ((long long)b*MAXSEQ + n0 + ty)*DM + colOff + c0 + tx;
#pragma unroll
  for (int i = 0; i < 32; i += 8) tile[ty+i][tx] = src[(long long)i*DM];
  __syncthreads();
  u16* dst = out + ((long long)b*outRows + c0 + ty)*MAXSEQ + n0 + tx;
#pragma unroll
  for (int i = 0; i < 32; i += 8) {
    u16 val = tile[tx][ty+i];
    if (cmax) {
      const int cc = c0 + ty + i;
      val = f2bf(expf(bf2f(val) - cmax[b*outRows + cc]) / csum[b*outRows + cc]);
    }
    dst[(long long)i*MAXSEQ] = val;
  }
}

// ---------------- local windowed attention (head 0) ----------------
// grid=(8 row-tiles, 32 windows, 8 batch), block=128. Writes ATTN cols [0,128).
__global__ void __launch_bounds__(128) local_attn_kernel(
    const u16* __restrict__ Qb, const u16* __restrict__ Kb,
    const u16* __restrict__ VT, u16* __restrict__ ATTN)
{
  __shared__ __align__(16) float sc[16][384];
  __shared__ __align__(16) u16  scb[16][384];
  const int b = blockIdx.z, win = blockIdx.y, mt = blockIdx.x;
  const int tid = threadIdx.x, w = tid >> 5, lane = tid & 31;
  const int half = lane >> 4, l16 = lane & 15;
  const int m0 = win*128 + mt*16;
  const long long qrow = ((long long)b*MAXSEQ + m0 + l16) * DM;

  v8f acc[6] = {};
  bool tm[6];
#pragma unroll
  for (int t = 0; t < 6; ++t) {
    const int ti = w*6 + t;
    tm[t] = (win == 0 && ti < 8) || (win == NWIN-1 && ti >= 16);
  }
  for (int k = 0; k < DH; k += 32) {
    FragBF a;
    a.q[0] = *(const uint4*)(Qb + qrow + k + half*8);
    a.q[1] = *(const uint4*)(Qb + qrow + k + 16 + half*8);
#pragma unroll
    for (int t = 0; t < 6; ++t) {
      if (tm[t]) continue;
      const int ti = w*6 + t;
      const int tokj = win*128 - 128 + ti*16 + l16;
      const u16* pb = Kb + ((long long)b*MAXSEQ + tokj)*DM;
      FragBF bf;
      bf.q[0] = *(const uint4*)(pb + k + half*8);
      bf.q[1] = *(const uint4*)(pb + k + 16 + half*8);
      acc[t] = WMMA_BF16(acc[t], a.v, bf.v);
    }
  }
  const float scale = 0.08838834764831845f;
#pragma unroll
  for (int t = 0; t < 6; ++t) {
    const int col = (w*6 + t)*16 + l16;
#pragma unroll
    for (int r = 0; r < 8; ++r)
      sc[r + 8*half][col] = tm[t] ? -INFINITY : acc[t][r]*scale;
  }
  __syncthreads();
  if (tid < 16) {
    float mx = -INFINITY;
    for (int j = 0; j < 384; ++j) mx = fmaxf(mx, sc[tid][j]);
    float s = 0.f;
    for (int j = 0; j < 384; ++j) { float e = expf(sc[tid][j] - mx); sc[tid][j] = e; s += e; }
    const float inv = 1.f / s;
    for (int j = 0; j < 384; ++j) scb[tid][j] = f2bf(sc[tid][j] * inv);
  }
  __syncthreads();

  v8f oacc[2] = {};
  for (int jb = 0; jb < 384; jb += 32) {
    FragBF a;
    a.q[0] = *(const uint4*)(&scb[l16][jb + half*8]);
    a.q[1] = *(const uint4*)(&scb[l16][jb + 16 + half*8]);
#pragma unroll
    for (int t = 0; t < 2; ++t) {
      const int d = (w*2 + t)*16 + l16;
      const u16* vrow = VT + ((long long)b*DM + d)*MAXSEQ;
      int t0 = win*128 - 128 + jb + half*8;
      int t1 = t0 + 16;
      t0 = min(max(t0, 0), MAXSEQ-8);
      t1 = min(max(t1, 0), MAXSEQ-8);
      FragBF bf;
      bf.q[0] = *(const uint4*)(vrow + t0);
      bf.q[1] = *(const uint4*)(vrow + t1);
      oacc[t] = WMMA_BF16(oacc[t], a.v, bf.v);
    }
  }
#pragma unroll
  for (int t = 0; t < 2; ++t) {
    const int d = (w*2 + t)*16 + l16;
#pragma unroll
    for (int r = 0; r < 8; ++r) {
      const int tok = m0 + r + 8*half;
      ATTN[((long long)b*MAXSEQ + tok)*DM + d] = f2bf(oacc[t][r]);
    }
  }
}

// ---------------- misc elementwise ----------------
__global__ void cvt_bf16_kernel(const float* __restrict__ in, u16* __restrict__ out, long n){
  long i = (long)blockIdx.x * 256 + threadIdx.x;
  if (i < n) out[i] = f2bf(in[i]);
}
__global__ void slice_out_kernel(const float* __restrict__ ws, float* __restrict__ out){
  long i = (long)blockIdx.x * 256 + threadIdx.x;     // over 8*3000*64
  if (i >= (long)NB*NTOK*DOUTN) return;
  int c = (int)(i % DOUTN); long t = i / DOUTN;
  int r = (int)(t % NTOK); int b = (int)(t / NTOK);
  out[i] = ws[((long long)b*MAXSEQ + r)*DOUTN + c];
}

// =======================================================================
extern "C" void kernel_launch(void* const* d_in, const int* in_sizes, int n_in,
                              void* d_out, int out_size, void* d_ws, size_t ws_size,
                              hipStream_t stream) {
  (void)in_sizes; (void)n_in; (void)out_size; (void)ws_size;
  const float* x    = (const float*)d_in[0];
  const float* W1   = (const float*)d_in[1];
  const float* b1   = (const float*)d_in[2];
  const float* ln1g = (const float*)d_in[3];
  const float* ln1b = (const float*)d_in[4];
  const float* Wq   = (const float*)d_in[5];
  const float* Wk   = (const float*)d_in[6];
  const float* Wv   = (const float*)d_in[7];
  const float* Wo   = (const float*)d_in[8];
  const float* bo   = (const float*)d_in[9];
  const float* ln2g = (const float*)d_in[10];
  const float* ln2b = (const float*)d_in[11];
  const float* Wf1  = (const float*)d_in[12];
  const float* bf1  = (const float*)d_in[13];
  const float* Wf2  = (const float*)d_in[14];
  const float* bf2  = (const float*)d_in[15];
  const float* W2   = (const float*)d_in[16];
  const float* b2   = (const float*)d_in[17];
  float* out = (float*)d_out;
  char* ws = (char*)d_ws;

  // ---- workspace layout (bytes) ----
  const size_t MB = 1024ull*1024ull;
  size_t oH    = 0;                 // H f32 [32768][512]              64MB
  size_t oXLN  = oH    + 64*MB;     // XLN bf16 [32768][512]           32MB (reused ln1/ln2)
  size_t oXb   = oXLN  + 32*MB;     // Xb bf16 [32768][128]             8MB
  size_t oW    = oXb   + 8*MB;      // weights bf16 (transposed)       <8MB
  size_t oW1t  = oW;
  size_t oWqt  = oW1t  + 131072;
  size_t oWkt  = oWqt  + 524288;
  size_t oWvt  = oWkt  + 524288;
  size_t oWot  = oWvt  + 524288;
  size_t oWf1t = oWot  + 524288;
  size_t oWf2t = oWf1t + 2097152;
  size_t oW2t  = oWf2t + 2097152;
  size_t oATTN = oW    + 8*MB;      // ATTN bf16 [32768][512]          32MB
  size_t oStat = oATTN + 32*MB;     // colmax/colsum f32 [24][128]
  size_t oCmax = oStat;
  size_t oCsum = oStat + 16384;
  size_t oCtx  = oStat + 65536;     // ctxT bf16 [24][128][128]        0.75MB
  size_t oOUT  = oCtx  + 1*MB;      // OUTWS f32 [32768][64]            8MB
  size_t oR    = oOUT  + 8*MB;      // big overlay region
  // attention phase layout inside region:
  size_t oQ    = oR;                // Q  bf16 [32768][512]            32MB
  size_t oK    = oR + 32*MB;        // K  bf16                         32MB
  size_t oV    = oR + 64*MB;        // V  bf16                         32MB
  size_t oQP   = oR + 96*MB;        // QP bf16 [32768][384]            24MB
  size_t oVT   = oR + 120*MB;       // VT bf16 [8][512][4096]          32MB
  size_t oKT2  = oR + 152*MB;       // KT2 bf16 [8][384][4096]         24MB
  // ffn phase overlay (attention buffers dead by then):
  size_t oFF1  = oR;                // FF1 bf16 [32768][2048]         128MB
  size_t oHb   = oR + 128*MB;       // Hb bf16 [32768][512]            32MB

  float* H    = (float*)(ws + oH);
  u16* XLN    = (u16*)(ws + oXLN);
  u16* Xb     = (u16*)(ws + oXb);
  u16* W1t    = (u16*)(ws + oW1t);
  u16* Wqt    = (u16*)(ws + oWqt);
  u16* Wkt    = (u16*)(ws + oWkt);
  u16* Wvt    = (u16*)(ws + oWvt);
  u16* Wot    = (u16*)(ws + oWot);
  u16* Wf1t   = (u16*)(ws + oWf1t);
  u16* Wf2t   = (u16*)(ws + oWf2t);
  u16* W2t    = (u16*)(ws + oW2t);
  u16* ATTN   = (u16*)(ws + oATTN);
  float* Cmax = (float*)(ws + oCmax);
  float* Csum = (float*)(ws + oCsum);
  u16* CtxT   = (u16*)(ws + oCtx);
  float* OUTW = (float*)(ws + oOUT);
  u16* Qb     = (u16*)(ws + oQ);
  u16* Kb     = (u16*)(ws + oK);
  u16* Vb     = (u16*)(ws + oV);
  u16* QP     = (u16*)(ws + oQP);
  u16* VT     = (u16*)(ws + oVT);
  u16* KT2    = (u16*)(ws + oKT2);
  u16* FF1    = (u16*)(ws + oFF1);
  u16* Hb     = (u16*)(ws + oHb);

  // ---- 1. weight conversion (f32 [K][N] -> bf16 [N][K]) ----
  wconv_kernel<<<(DIN*DM+255)/256,  256, 0, stream>>>(W1,  W1t,  DIN, DM);
  wconv_kernel<<<(DM*DM+255)/256,   256, 0, stream>>>(Wq,  Wqt,  DM,  DM);
  wconv_kernel<<<(DM*DM+255)/256,   256, 0, stream>>>(Wk,  Wkt,  DM,  DM);
  wconv_kernel<<<(DM*DM+255)/256,   256, 0, stream>>>(Wv,  Wvt,  DM,  DM);
  wconv_kernel<<<(DM*DM+255)/256,   256, 0, stream>>>(Wo,  Wot,  DM,  DM);
  wconv_kernel<<<(DM*DFF+255)/256,  256, 0, stream>>>(Wf1, Wf1t, DM,  DFF);
  wconv_kernel<<<(DFF*DM+255)/256,  256, 0, stream>>>(Wf2, Wf2t, DFF, DM);
  wconv_kernel<<<(DM*DOUTN+255)/256,256, 0, stream>>>(W2,  W2t,  DM,  DOUTN);

  // ---- 2. pad & convert x ----
  pad_x_kernel<<<((long)ROWS*DIN+255)/256, 256, 0, stream>>>(x, Xb);

  // ---- 3. H = tanh(Xb @ W1 + b1) ----
  gemm_wmma<<<dim3(ROWS/128, DM/64, 1), 128, 0, stream>>>(
      Xb, 0, 0, DIN, W1t, 0, 0, DIN, H, 0, 0, DM, b1, ROWS, DM, DIN, 1, 4);

  // ---- 4. XLN = LN1(H) ----
  ln_kernel<<<ROWS/8, 256, 0, stream>>>(H, ln1g, ln1b, XLN);

  // ---- 5. Q,K,V = XLN @ Wq/Wk/Wv (bf16) ----
  gemm_wmma<<<dim3(ROWS/128, DM/64, 1), 128, 0, stream>>>(
      XLN, 0, 0, DM, Wqt, 0, 0, DM, Qb, 0, 0, DM, nullptr, ROWS, DM, DM, 1, 1);
  gemm_wmma<<<dim3(ROWS/128, DM/64, 1), 128, 0, stream>>>(
      XLN, 0, 0, DM, Wkt, 0, 0, DM, Kb, 0, 0, DM, nullptr, ROWS, DM, DM, 1, 1);
  gemm_wmma<<<dim3(ROWS/128, DM/64, 1), 128, 0, stream>>>(
      XLN, 0, 0, DM, Wvt, 0, 0, DM, Vb, 0, 0, DM, nullptr, ROWS, DM, DM, 1, 1);

  // ---- 6. linear-attn prep: q row-softmax, k column stats, transposes ----
  qprep_kernel<<<ROWS, 96, 0, stream>>>(Qb, QP);
  kstats_kernel<<<NB*3, 256, 0, stream>>>(Kb, Cmax, Csum);
  transpose_kernel<<<dim3(MAXSEQ/32, DM/32, NB), 256, 0, stream>>>(
      Vb, VT, nullptr, nullptr, 0, DM);               // VT[b][512][4096]
  transpose_kernel<<<dim3(MAXSEQ/32, 384/32, NB), 256, 0, stream>>>(
      Kb, KT2, Cmax, Csum, DH, 384);                  // KT2[b][384][4096] = k'' (norm)

  // ---- 7. ctxT[e][d] = v^T @ k''  (batched over 8*3) ----
  gemm_wmma<<<dim3(1, DH/64, NB*3), 128, 0, stream>>>(
      VT + (long long)DH*MAXSEQ, (long long)DM*MAXSEQ, (long long)DH*MAXSEQ, MAXSEQ,
      KT2, (long long)384*MAXSEQ, (long long)DH*MAXSEQ, MAXSEQ,
      CtxT, (long long)3*DH*DH, (long long)DH*DH, DH,
      nullptr, DH, DH, MAXSEQ, 3, 1);

  // ---- 8. local windowed attention (head 0) -> ATTN[:, 0:128] ----
  local_attn_kernel<<<dim3(8, NWIN, NB), 128, 0, stream>>>(Qb, Kb, VT, ATTN);

  // ---- 9. global heads: out = q' @ ctx -> ATTN[:, 128:512] ----
  gemm_wmma<<<dim3(MAXSEQ/128, DH/64, NB*3), 128, 0, stream>>>(
      QP, (long long)MAXSEQ*384, (long long)DH, 384,
      CtxT, (long long)3*DH*DH, (long long)DH*DH, DH,
      ATTN + DH, (long long)MAXSEQ*DM, (long long)DH, DM,
      nullptr, MAXSEQ, DH, DH, 3, 1);

  // ---- 10. H += ATTN @ Wo + bo ----
  gemm_wmma<<<dim3(ROWS/128, DM/64, 1), 128, 0, stream>>>(
      ATTN, 0, 0, DM, Wot, 0, 0, DM, H, 0, 0, DM, bo, ROWS, DM, DM, 1, 2);

  // ---- 11. XLN = LN2(H) ----
  ln_kernel<<<ROWS/8, 256, 0, stream>>>(H, ln2g, ln2b, XLN);

  // ---- 12. FF1 = gelu(XLN @ Wf1 + bf1) (bf16) ----
  gemm_wmma<<<dim3(ROWS/128, DFF/64, 1), 128, 0, stream>>>(
      XLN, 0, 0, DM, Wf1t, 0, 0, DM, FF1, 0, 0, DFF, bf1, ROWS, DFF, DM, 1, 1|8);

  // ---- 13. H += FF1 @ Wf2 + bf2 ----
  gemm_wmma<<<dim3(ROWS/128, DM/64, 1), 128, 0, stream>>>(
      FF1, 0, 0, DFF, Wf2t, 0, 0, DFF, H, 0, 0, DM, bf2, ROWS, DM, DFF, 1, 2);

  // ---- 14. OUTW = H @ W2 + b2 ----
  cvt_bf16_kernel<<<((long)ROWS*DM+255)/256, 256, 0, stream>>>(H, Hb, (long)ROWS*DM);
  gemm_wmma<<<dim3(ROWS/128, DOUTN/64, 1), 128, 0, stream>>>(
      Hb, 0, 0, DM, W2t, 0, 0, DM, OUTW, 0, 0, DOUTN, b2, ROWS, DOUTN, DM, 1, 0);

  // ---- 15. slice to [8,3000,64] ----
  slice_out_kernel<<<((long)NB*NTOK*DOUTN+255)/256, 256, 0, stream>>>(OUTW, out);
}
